// ClusteringModule_41918880809336
// MI455X (gfx1250) — matrix operand order repaired and checked by reading, not compiled, not run on hardware
//
#include <hip/hip_runtime.h>
#include <hip/hip_bf16.h>
#include <math.h>

// ---------------- problem constants (from reference) ----------------
#define BNB   16
#define SNB   64
#define CCH   480
#define TT    8
#define FYD   28
#define FXD   28
#define OSD   8
#define NDF   256
#define KCLUS 64
#define GK    (CCH*OSD*OSD)     // 30720  (GEMM K)
#define MTOT  (BNB*SNB)         // 1024   (GEMM M)
#define MT    32                // M tile per block
#define KT    32                // K step
#define NSTEP (GK/KT)           // 960
#define IMGW  1920.0f
#define IMGH  1080.0f

// ---------------- CDNA5 WMMA types ----------------
typedef __attribute__((ext_vector_type(16))) __bf16 v16bf;
typedef __attribute__((ext_vector_type(8)))  float  v8f;
typedef __attribute__((ext_vector_type(4)))  int    v4i;

union FragAB { v16bf v; unsigned u[8]; };

// gfx1250 async global->LDS copy (ASYNCcnt) if this toolchain declares it
#if defined(__has_builtin)
#  if __has_builtin(__builtin_amdgcn_global_load_async_to_lds_b128)
#    define HAVE_ASYNC_LDS 1
#  endif
#  if __has_builtin(__builtin_amdgcn_s_wait_asynccnt)
#    define HAVE_WAIT_ASYNC 1
#  endif
#endif

// exact parameter types per hipcc diagnostic: v4i in AS1 (global) / AS3 (LDS)
typedef __attribute__((address_space(1))) v4i gv4i_t;
typedef __attribute__((address_space(3))) v4i sv4i_t;

static __device__ __forceinline__ void wait_async0() {
#if defined(HAVE_ASYNC_LDS)
#  if defined(HAVE_WAIT_ASYNC)
    __builtin_amdgcn_s_wait_asynccnt(0);
#  else
    asm volatile("s_wait_asynccnt 0x0" ::: "memory");
#  endif
#endif
}

// fp32 -> bf16, round to nearest even
static __device__ __forceinline__ unsigned short f2bf(float f) {
    unsigned u = __builtin_bit_cast(unsigned, f);
    u = (u + 0x7FFFu + ((u >> 16) & 1u)) >> 16;
    return (unsigned short)u;
}

// ---------------- pre-pass: W_vis fp32 [K][N] -> bf16 tiles [K/32][N][32] ----------------
// One-shot: hoists 960x8192 per-block fp32->bf16 conversions out of the GEMM hot loop and
// makes each 32xK B tile a single contiguous 16KB block (async-copy friendly).
__global__ __launch_bounds__(256)
void convert_W_kernel(const float* __restrict__ W_vis, unsigned short* __restrict__ Wb) {
    int e  = blockIdx.x * 256 + threadIdx.x;       // over GK * (NDF/4) = 30720*64
    int k  = e >> 6;
    int n4 = (e & 63) << 2;
    float4 w = *reinterpret_cast<const float4*>(W_vis + (size_t)k*NDF + n4);
    unsigned short* dst = Wb + ((size_t)(k >> 5) * (NDF*KT)) + (k & 31);
    dst[(size_t)(n4+0)*KT] = f2bf(w.x);
    dst[(size_t)(n4+1)*KT] = f2bf(w.y);
    dst[(size_t)(n4+2)*KT] = f2bf(w.z);
    dst[(size_t)(n4+3)*KT] = f2bf(w.w);
}

// ---------------- fused main kernel ----------------
// RoIAlign -> (roi @ W, bf16 WMMA, fp32 accum, double-buffered async staging) ->
// +bias/spatial -> centroid distances -> student-t soft assignment -> argmax.
// Grid: 32 blocks (M tiles of 32 ROIs), 256 threads (8 wave32).
__global__ __launch_bounds__(256)
void cluster_fused_kernel(const float* __restrict__ z_vis,
                          const float* __restrict__ bboxs,
                          const unsigned short* __restrict__ Wb,
                          const float* __restrict__ b_vis,
                          const float* __restrict__ W_spc,
                          const float* __restrict__ b_spc,
                          const float* __restrict__ centroids,
                          float* __restrict__ out_z,
                          float* __restrict__ out_s,
                          float* __restrict__ out_c)
{
    // ----- LDS (~111 KB) -----
    __shared__ int            g_off[MT*64];
    __shared__ int            g_dxy[MT*64];
    __shared__ float          g_lx [MT*64];
    __shared__ float          g_ly [MT*64];
    __shared__ float          g_sc [MT*64];
    __shared__ float          norms_s[MT];
    __shared__ unsigned short A_lds[2][MT*KT];     // double-buffered [r][k] bf16
    __shared__ unsigned short B_lds[2][NDF*KT];    // double-buffered [n][k] bf16
    __shared__ float          z_lds[MT*NDF];
    __shared__ float          red_sum[MT*8];
    __shared__ float          red_max[MT*8];
    __shared__ int            red_arg[MT*8];

    const int tid = threadIdx.x;
    const int bm  = blockIdx.x;
    const int m0  = bm * MT;
    const int b   = m0 / SNB;               // batch, constant per block

    // ----- RoIAlign geometry precompute (bins repeat every channel) -----
    const float sx = (float)FXD / IMGW, sy = (float)FYD / IMGH;
    for (int i = 0; i < 8; ++i) {
        int e   = tid + i*256;
        int r   = e >> 6;
        int bin = e & 63;
        int m   = m0 + r;
        const float* bx = bboxs + (size_t)((b*SNB + (m & (SNB-1))) << 2);
        float x1 = bx[0]*sx - 0.5f, y1 = bx[1]*sy - 0.5f;
        float x2 = bx[2]*sx - 0.5f, y2 = bx[3]*sy - 0.5f;
        float bw = (x2 - x1) * (1.0f/OSD), bh = (y2 - y1) * (1.0f/OSD);
        int  px = bin & 7, py = bin >> 3;
        float X = x1 + ((float)px + 0.5f)*bw;
        float Y = y1 + ((float)py + 0.5f)*bh;
        bool valid = (Y > -1.0f) && (Y < (float)FYD) && (X > -1.0f) && (X < (float)FXD);
        float x = fminf(fmaxf(X, 0.0f), (float)(FXD-1));
        float y = fminf(fmaxf(Y, 0.0f), (float)(FYD-1));
        int x0 = (int)floorf(x), y0 = (int)floorf(y);
        g_off[e] = y0*FXD + x0;
        int dx = (x0 < FXD-1) ? 1   : 0;
        int dy = (y0 < FYD-1) ? FXD : 0;
        g_dxy[e] = dx | (dy << 16);
        g_lx[e]  = x - (float)x0;
        g_ly[e]  = y - (float)y0;
        g_sc[e]  = valid ? 1.0f : 0.0f;
    }
    if (tid < MT) {
        int m = m0 + tid;
        const float* bx = bboxs + (size_t)((b*SNB + (m & (SNB-1))) << 2);
        float cx = (bx[0] + bx[2])*0.5f - IMGW*0.5f;
        float cy = (bx[1] + bx[3])*0.5f - IMGH*0.5f;
        norms_s[tid] = sqrtf(cx*cx + cy*cy) *
                       (1.0f / sqrtf(IMGW*IMGW*0.25f + IMGH*IMGH*0.25f));
    }

    // ----- wave/tile mapping -----
    const int lane = tid & 31;
    const int wave = tid >> 5;
    const int sel  = lane >> 4;
    const int rc   = lane & 15;
    const int mt   = wave & 1;
    const int ntb  = (wave >> 1) * 4;

    v8f acc[4] = {};

    // stage step `s` into buffer `bs`: RoIAlign-gather A tile + async-copy B tile
    auto stage = [&](int s, int bs) {
        int kk      = s * KT;
        int cc      = kk >> 6;
        int binbase = kk & 63;
        const float* featc =
            z_vis + (((size_t)(b*CCH + cc)*TT + (TT-1)) * (size_t)(FYD*FXD));
        unsigned short* Ab = &A_lds[bs][0];
        #pragma unroll
        for (int i = 0; i < 4; ++i) {
            int e = tid + i*256;
            int r = e >> 5, t = e & 31;
            int gi  = r*64 + binbase + t;
            int off = g_off[gi];
            int dxy = g_dxy[gi];
            int dx  = dxy & 0xFFFF, dy = dxy >> 16;
            float lx = g_lx[gi], ly = g_ly[gi];
            float v00 = featc[off],      v01 = featc[off + dx];
            float v10 = featc[off + dy], v11 = featc[off + dy + dx];
            float val = g_sc[gi] * ((1.0f-ly)*((1.0f-lx)*v00 + lx*v01)
                                   +       ly *((1.0f-lx)*v10 + lx*v11));
            Ab[r*KT + t] = f2bf(val);
        }
        // B tile: contiguous 16KB block Wb[s] -> LDS (256 thr x 4 x 16B)
        const char* gsrc = (const char*)Wb + ((size_t)s << 14) + (size_t)tid*16;
        char*       ldst = (char*)&B_lds[bs][0] + tid*16;
#if defined(HAVE_ASYNC_LDS)
        __builtin_amdgcn_global_load_async_to_lds_b128((gv4i_t*)(gsrc        ), (sv4i_t*)(ldst        ), 0, 0);
        __builtin_amdgcn_global_load_async_to_lds_b128((gv4i_t*)(gsrc +  4096), (sv4i_t*)(ldst +  4096), 0, 0);
        __builtin_amdgcn_global_load_async_to_lds_b128((gv4i_t*)(gsrc +  8192), (sv4i_t*)(ldst +  8192), 0, 0);
        __builtin_amdgcn_global_load_async_to_lds_b128((gv4i_t*)(gsrc + 12288), (sv4i_t*)(ldst + 12288), 0, 0);
#else
        const uint4* g = reinterpret_cast<const uint4*>(gsrc);
        uint4*       l = reinterpret_cast<uint4*>(ldst);
        l[0]   = g[0];
        l[256] = g[256];
        l[512] = g[512];
        l[768] = g[768];
#endif
    };

    // ----- software-pipelined main loop -----
    stage(0, 0);
    for (int i = 0; i < NSTEP; ++i) {
        int cur = i & 1;
        wait_async0();            // stage(i) async copies complete (issued last iter)
        __syncthreads();          // stage(i) LDS writes visible; buf[cur^1] readers done
        if (i + 1 < NSTEP) stage(i + 1, cur ^ 1);   // overlap with compute(i)

        FragAB a;
        #pragma unroll
        for (int p = 0; p < 8; ++p) {
            int kb = ((p < 4) ? 2*p : 2*p + 8) + sel*8;
            a.u[p] = *reinterpret_cast<const unsigned*>(&A_lds[cur][(mt*16 + rc)*KT + kb]);
        }
        #pragma unroll
        for (int j = 0; j < 4; ++j) {
            FragAB bf;
            int n = (ntb + j)*16 + rc;
            #pragma unroll
            for (int p = 0; p < 8; ++p) {
                int kb = ((p < 4) ? 2*p : 2*p + 8) + sel*8;
                bf.u[p] = *reinterpret_cast<const unsigned*>(&B_lds[cur][n*KT + kb]);
            }
            acc[j] = __builtin_amdgcn_wmma_f32_16x16x32_bf16(
                         false, a.v, false, bf.v, (short)0, acc[j], false, false);
        }
    }

    __syncthreads();

    // ----- epilogue 1: z = acc + b_vis + norms*W_spc + b_spc -----
    #pragma unroll
    for (int j = 0; j < 4; ++j) {
        int n = (ntb + j)*16 + rc;
        float bias = b_vis[n] + b_spc[n];
        float wsp  = W_spc[n];
        #pragma unroll
        for (int r = 0; r < 8; ++r) {
            int   ml = mt*16 + sel*8 + r;
            float zv = acc[j][r] + bias + norms_s[ml]*wsp;
            z_lds[ml*NDF + n]                = zv;
            out_z[(size_t)(m0 + ml)*NDF + n] = zv;
        }
    }
    __syncthreads();

    // ----- epilogue 2: student-t soft assignment over 64 centroids -----
    {
        int ml = tid >> 3;
        int q  = tid & 7;
        const float* zr = &z_lds[ml*NDF];
        float su[8];
        float ssum = 0.0f, bmax = -1.0f;
        int   barg = 0;
        for (int jj = 0; jj < 8; ++jj) {
            int cl = q*8 + jj;
            const float* cv = centroids + (size_t)cl*NDF;
            float a2 = 0.0f;
            #pragma unroll 8
            for (int d = 0; d < NDF; ++d) {
                float df = zr[d] - cv[d];
                a2 = fmaf(df, df, a2);
            }
            float dist = sqrtf(a2);
            float sv   = 1.0f / (1.0f + dist);    // ALPHA=1
            su[jj] = sv;
            ssum  += sv;
            if (sv > bmax) { bmax = sv; barg = cl; }
        }
        red_sum[ml*8 + q] = ssum;
        red_max[ml*8 + q] = bmax;
        red_arg[ml*8 + q] = barg;
        __syncthreads();

        float tot = 0.0f;
        for (int qq = 0; qq < 8; ++qq) tot += red_sum[ml*8 + qq];
        float inv = 1.0f / tot;
        int m = m0 + ml;
        for (int jj = 0; jj < 8; ++jj)
            out_s[(size_t)m*KCLUS + q*8 + jj] = su[jj] * inv;

        if (q == 0) {
            float bv = red_max[ml*8]; int ba = red_arg[ml*8];
            for (int qq = 1; qq < 8; ++qq)
                if (red_max[ml*8 + qq] > bv) { bv = red_max[ml*8 + qq]; ba = red_arg[ml*8 + qq]; }
            out_c[m] = (float)ba;
        }
    }
}

extern "C" void kernel_launch(void* const* d_in, const int* in_sizes, int n_in,
                              void* d_out, int out_size, void* d_ws, size_t ws_size,
                              hipStream_t stream) {
    (void)in_sizes; (void)n_in; (void)out_size; (void)ws_size;
    const float* z_vis     = (const float*)d_in[0];
    const float* bboxs     = (const float*)d_in[1];
    const float* W_vis     = (const float*)d_in[2];
    const float* b_vis     = (const float*)d_in[3];
    const float* W_spc     = (const float*)d_in[4];
    const float* b_spc     = (const float*)d_in[5];
    const float* centroids = (const float*)d_in[6];

    unsigned short* Wb = (unsigned short*)d_ws;     // GK*NDF bf16 = 15.7 MB

    float* o     = (float*)d_out;
    float* out_z = o;
    float* out_s = o + (size_t)MTOT*NDF;
    float* out_c = o + (size_t)MTOT*NDF + (size_t)MTOT*KCLUS;

    convert_W_kernel<<<dim3(GK*(NDF/4)/256), dim3(256), 0, stream>>>(W_vis, Wb);
    cluster_fused_kernel<<<dim3(MTOT/MT), dim3(256), 0, stream>>>(
        z_vis, bboxs, Wb, b_vis, W_spc, b_spc, centroids,
        out_z, out_s, out_c);
}
